// SpatialGNN_51333449122159
// MI455X (gfx1250) — matrix operand order repaired and checked by reading, not compiled, not run on hardware
//
#include <hip/hip_runtime.h>
#include <hip/hip_bf16.h>
#include <math.h>

// Problem constants (from reference)
#define BB     64
#define NN     1024      // 32x32 grid
#define DFEAT  1024
#define CTX    512
#define SS     32
#define TT     4
#define MROWS  (BB * NN) // 65536

typedef __attribute__((ext_vector_type(16))) __bf16 v16bf;
typedef __attribute__((ext_vector_type(8)))  float  v8f;

// ---------------------------------------------------------------------------
// A-fragment (16x32 bf16, MxK): lane L: row = row0 + (L&15), hs = L>>4.
// elements 0..7  <-> K = k0 + 8*hs      + 0..7
// elements 8..15 <-> K = k0 + 16 + 8*hs + 0..7
// ---------------------------------------------------------------------------
__device__ __forceinline__ v16bf load_fragA(const float* __restrict__ row,
                                            int k0, int hs) {
  const float4* p0 = reinterpret_cast<const float4*>(row + k0 + 8 * hs);
  const float4* p1 = reinterpret_cast<const float4*>(row + k0 + 16 + 8 * hs);
  float4 a = p0[0], b = p0[1], c = p1[0], d = p1[1];
  v16bf f;
  f[0]  = (__bf16)a.x; f[1]  = (__bf16)a.y; f[2]  = (__bf16)a.z; f[3]  = (__bf16)a.w;
  f[4]  = (__bf16)b.x; f[5]  = (__bf16)b.y; f[6]  = (__bf16)b.z; f[7]  = (__bf16)b.w;
  f[8]  = (__bf16)c.x; f[9]  = (__bf16)c.y; f[10] = (__bf16)c.z; f[11] = (__bf16)c.w;
  f[12] = (__bf16)d.x; f[13] = (__bf16)d.y; f[14] = (__bf16)d.z; f[15] = (__bf16)d.w;
  return f;
}

// ---------------------------------------------------------------------------
// B-fragment (32x16 bf16, KxN): lane L: col = col0 + (L&15), hs = L>>4.
// elements 0..7 <-> K = k0 + 8*hs + 0..7, elements 8..15 <-> K = k0 + 16 + 8*hs + 0..7.
// W stored row-major [Nout, K] -> two contiguous 8-float runs per lane.
// ---------------------------------------------------------------------------
__device__ __forceinline__ v16bf load_fragB(const float* __restrict__ row,
                                            int k0, int hs) {
  const float4* p0 = reinterpret_cast<const float4*>(row + k0 + 8 * hs);
  const float4* p1 = reinterpret_cast<const float4*>(row + k0 + 16 + 8 * hs);
  float4 a = p0[0], b = p0[1], c = p1[0], d = p1[1];
  v16bf f;
  f[0]  = (__bf16)a.x; f[1]  = (__bf16)a.y; f[2]  = (__bf16)a.z; f[3]  = (__bf16)a.w;
  f[4]  = (__bf16)b.x; f[5]  = (__bf16)b.y; f[6]  = (__bf16)b.z; f[7]  = (__bf16)b.w;
  f[8]  = (__bf16)c.x; f[9]  = (__bf16)c.y; f[10] = (__bf16)c.z; f[11] = (__bf16)c.w;
  f[12] = (__bf16)d.x; f[13] = (__bf16)d.y; f[14] = (__bf16)d.z; f[15] = (__bf16)d.w;
  return f;
}

// ---------------------------------------------------------------------------
// WMMA GEMM: C[row, col] = sum_k A[row,k] * W[col,k] + bias
// A: [MROWS, lda] fp32, W: [512, ldb] fp32 (row-major over K), C: [MROWS, 512]
// biasMode 0: bias[col] ; biasMode 1: bias[(row>>10)*512 + col]  (per-batch)
// Block tile: 32 rows x 512 cols, 8 waves: wave&1 = row half (16 rows),
// wave>>1 = 128-col group (8 WMMA tiles). grid = MROWS/32 = 2048 blocks.
// Inner loop: stage all 8 B-fragments in distinct registers, then fire the
// 8 WMMAs back-to-back (disjoint D/B regs -> no intra-burst hazard NOPs;
// the WAR on fragment regs occurs once per K-step and is covered by the
// next iteration's loads/address math).
// ---------------------------------------------------------------------------
__global__ __launch_bounds__(256) void gemm_wmma_bf16(
    const float* __restrict__ A, const float* __restrict__ W,
    float* __restrict__ C, int K, int lda, int ldb,
    const float* __restrict__ bias, int biasMode) {
  const int wave = threadIdx.x >> 5;
  const int lane = threadIdx.x & 31;
  const int hs   = lane >> 4;
  const int l15  = lane & 15;

  const int row0 = blockIdx.x * 32 + (wave & 1) * 16;
  const int col0 = (wave >> 1) * 128;

  const float* Arow = A + (size_t)(row0 + l15) * lda;
  const float* Brow[8];
#pragma unroll
  for (int t = 0; t < 8; ++t)
    Brow[t] = W + (size_t)(col0 + t * 16 + l15) * ldb;

  v8f acc[8] = {};

  for (int k0 = 0; k0 < K; k0 += 32) {
    __builtin_prefetch(Arow + k0 + 128, 0, 1);  // global_prefetch_b8 on A stream

    v16bf af = load_fragA(Arow, k0, hs);
    v16bf bf[8];
#pragma unroll
    for (int t = 0; t < 8; ++t) bf[t] = load_fragB(Brow[t], k0, hs);

#pragma unroll
    for (int t = 0; t < 8; ++t) {
      acc[t] = __builtin_amdgcn_wmma_f32_16x16x32_bf16(
          /*neg_a=*/false, af, /*neg_b=*/false, bf[t],
          /*c_mod=*/(short)0, acc[t], /*reuse_a=*/false, /*reuse_b=*/false);
    }
  }

  // Store: element e of acc -> row = row0 + e + 8*hs, col = col0 + t*16 + l15
  const int bat = row0 >> 10;
#pragma unroll
  for (int t = 0; t < 8; ++t) {
    const int col = col0 + t * 16 + l15;
    const float bv = biasMode ? bias[bat * CTX + col] : bias[col];
    float* Cp = C + (size_t)row0 * CTX + col;
#pragma unroll
    for (int e = 0; e < 8; ++e) {
      Cp[(size_t)(e + 8 * hs) * CTX] = acc[t][e] + bv;
    }
  }
}

// ---------------------------------------------------------------------------
// q1 = elu(q_encoding @ qInput_w^T + qInput_b)   [64, 512]
// ---------------------------------------------------------------------------
__global__ __launch_bounds__(256) void q1_kernel(
    const float* __restrict__ qe, const float* __restrict__ Wq,
    const float* __restrict__ bq, float* __restrict__ q1out) {
  const int b = blockIdx.x;
  __shared__ float qs[CTX];
  for (int i = threadIdx.x; i < CTX; i += 256) qs[i] = qe[b * CTX + i];
  __syncthreads();
  for (int c = threadIdx.x; c < CTX; c += 256) {
    float s = bq[c];
    const float* w = Wq + (size_t)c * CTX;
    for (int k = 0; k < CTX; ++k) s = fmaf(qs[k], w[k], s);
    q1out[b * CTX + c] = s > 0.f ? s : (__expf(s) - 1.f);
  }
}

// ---------------------------------------------------------------------------
// Build ≤8-entry neighbor lists per row of the dense adjacency (run once).
// ---------------------------------------------------------------------------
__global__ void nbr_kernel(const float* __restrict__ adj, int* __restrict__ nidx,
                           float* __restrict__ nw, int* __restrict__ ncnt) {
  const int n = blockIdx.x * blockDim.x + threadIdx.x;
  if (n >= NN) return;
  int cnt = 0;
  for (int m = 0; m < NN; ++m) {
    float a = adj[(size_t)n * NN + m];
    if (a != 0.f && cnt < 8) { nidx[n * 8 + cnt] = m; nw[n * 8 + cnt] = a; ++cnt; }
  }
  ncnt[n] = cnt;
}

// ---------------------------------------------------------------------------
// Per-iteration command pipeline (one block per batch element):
//   q_cmd -> masked softmax attention over S -> cmd -> cmd2 = cmd@W2^T + W_b
// ---------------------------------------------------------------------------
__global__ __launch_bounds__(256) void cmd_kernel(
    const float* __restrict__ q1, const float* __restrict__ lstm,
    const int* __restrict__ qlen, const float* __restrict__ WqT,
    const float* __restrict__ bqT, const float* __restrict__ cmdw,
    const float* __restrict__ cmdb, const float* __restrict__ Ww,
    const float* __restrict__ Wb, float* __restrict__ cmd2out) {
  const int b = blockIdx.x;
  __shared__ float q1s[CTX];
  __shared__ float qc2[CTX];
  __shared__ float red[256];
  __shared__ float att[SS];
  __shared__ float cmds[CTX];
  const float* lb = lstm + (size_t)b * SS * CTX;

  for (int i = threadIdx.x; i < CTX; i += 256) q1s[i] = q1[b * CTX + i];
  __syncthreads();

  // q_cmd[c] * cmd_w[c]
  for (int c = threadIdx.x; c < CTX; c += 256) {
    float s = bqT[c];
    const float* w = WqT + (size_t)c * CTX;
    for (int k = 0; k < CTX; ++k) s = fmaf(q1s[k], w[k], s);
    qc2[c] = s * cmdw[c];
  }
  __syncthreads();

  // raw attention: 8 threads per s
  {
    const int s = threadIdx.x >> 3, g = threadIdx.x & 7;
    float p = 0.f;
    const float* lr = lb + (size_t)s * CTX;
    for (int d = g; d < CTX; d += 8) p = fmaf(qc2[d], lr[d], p);
    red[threadIdx.x] = p;
  }
  __syncthreads();
  if ((threadIdx.x & 7) == 0) {
    const int s = threadIdx.x >> 3;
    float r = 0.f;
    for (int g = 0; g < 8; ++g) r += red[(s << 3) + g];
    r += cmdb[0];
    if (s >= qlen[b]) r = -1e30f;
    att[s] = r;
  }
  __syncthreads();
  if (threadIdx.x == 0) {  // masked stable softmax over S=32
    float mx = att[0];
    for (int s = 1; s < SS; ++s) mx = fmaxf(mx, att[s]);
    float sum = 0.f;
    for (int s = 0; s < SS; ++s) { float e = __expf(att[s] - mx); att[s] = e; sum += e; }
    float inv = 1.f / sum;
    for (int s = 0; s < SS; ++s) att[s] *= inv;
  }
  __syncthreads();

  for (int d = threadIdx.x; d < CTX; d += 256) {
    float c = 0.f;
    for (int s = 0; s < SS; ++s) c = fmaf(att[s], lb[(size_t)s * CTX + d], c);
    cmds[d] = c;
  }
  __syncthreads();

  // cmd2[c] = W_b[c] + sum_d cmds[d] * W_w[c, 512+d]   (the W2 half of concat)
  for (int c = threadIdx.x; c < CTX; c += 256) {
    float s = Wb[c];
    const float* w = Ww + (size_t)c * (2 * CTX) + CTX;
    for (int d = 0; d < CTX; ++d) s = fmaf(cmds[d], w[d], s);
    cmd2out[b * CTX + c] = s;
  }
}

// ---------------------------------------------------------------------------
// Sparse adjacency apply: x_new[b,n,:] = sum_j w_j * support[b, m_j, :]
// + optional ELU. One block per (b,n); each thread covers 2 channels (float2).
// ---------------------------------------------------------------------------
__global__ __launch_bounds__(256) void adj_apply(
    const float* __restrict__ sup, const int* __restrict__ nidx,
    const float* __restrict__ nw, const int* __restrict__ ncnt,
    float* __restrict__ out, int doElu) {
  const int bn = blockIdx.x;
  const int b = bn >> 10, n = bn & (NN - 1);
  const int cnt = ncnt[n];
  float2 acc = make_float2(0.f, 0.f);
  for (int j = 0; j < cnt; ++j) {
    const int m = nidx[n * 8 + j];
    const float w = nw[n * 8 + j];
    const float2 v = reinterpret_cast<const float2*>(
        sup + (((size_t)b << 10) + m) * CTX)[threadIdx.x];
    acc.x = fmaf(w, v.x, acc.x);
    acc.y = fmaf(w, v.y, acc.y);
  }
  if (doElu) {
    acc.x = acc.x > 0.f ? acc.x : (__expf(acc.x) - 1.f);
    acc.y = acc.y > 0.f ? acc.y : (__expf(acc.y) - 1.f);
  }
  reinterpret_cast<float2*>(out + (((size_t)b << 10) + n) * CTX)[threadIdx.x] = acc;
}

// ---------------------------------------------------------------------------
extern "C" void kernel_launch(void* const* d_in, const int* in_sizes, int n_in,
                              void* d_out, int out_size, void* d_ws, size_t ws_size,
                              hipStream_t stream) {
  const float* images   = (const float*)d_in[0];
  const float* qenc     = (const float*)d_in[1];
  const float* lstm     = (const float*)d_in[2];
  const int*   qlen     = (const int*)  d_in[3];
  const float* adj      = (const float*)d_in[4];
  const float* initKB_w = (const float*)d_in[5];
  const float* initKB_b = (const float*)d_in[6];
  const float* Ww       = (const float*)d_in[7];
  const float* Wb       = (const float*)d_in[8];
  const float* qIw      = (const float*)d_in[9];
  const float* qIb      = (const float*)d_in[10];
  const float* qTw      = (const float*)d_in[11];
  const float* qTb      = (const float*)d_in[12];
  const float* cmdw     = (const float*)d_in[13];
  const float* cmdb     = (const float*)d_in[14];

  float* xloc = (float*)d_out;  // [B,N,CTX] ping buffer (also final output)

  // Workspace layout
  float* ws_sup  = (float*)d_ws;                       // 64*1024*512 floats
  float* ws_q1   = ws_sup + (size_t)MROWS * CTX;       // 64*512
  float* ws_cmd2 = ws_q1 + BB * CTX;                   // 64*512
  int*   ws_nidx = (int*)(ws_cmd2 + BB * CTX);         // 1024*8
  float* ws_nw   = (float*)(ws_nidx + NN * 8);         // 1024*8
  int*   ws_ncnt = (int*)(ws_nw + NN * 8);             // 1024

  // One-time prep
  q1_kernel<<<BB, 256, 0, stream>>>(qenc, qIw, qIb, ws_q1);
  nbr_kernel<<<(NN + 255) / 256, 256, 0, stream>>>(adj, ws_nidx, ws_nw, ws_ncnt);

  // x_loc = images @ initKB_w^T + initKB_b   (K=1024)
  gemm_wmma_bf16<<<MROWS / 32, 256, 0, stream>>>(
      images, initKB_w, xloc, DFEAT, DFEAT, DFEAT, initKB_b, 0);

  for (int t = 0; t < TT; ++t) {
    cmd_kernel<<<BB, 256, 0, stream>>>(
        ws_q1, lstm, qlen, qTw + (size_t)t * CTX * CTX, qTb + (size_t)t * CTX,
        cmdw, cmdb, Ww, Wb, ws_cmd2);

    // support = x_loc @ W1^T + cmd2[b]   (K=512, W row stride 1024)
    gemm_wmma_bf16<<<MROWS / 32, 256, 0, stream>>>(
        xloc, Ww, ws_sup, CTX, CTX, 2 * CTX, ws_cmd2, 1);

    // x_loc = adj(sparse) @ support, ELU except last iteration
    adj_apply<<<MROWS, 256, 0, stream>>>(
        ws_sup, ws_nidx, ws_nw, ws_ncnt, xloc, (t < TT - 1) ? 1 : 0);
  }
}
